// RuleRNN_11003706213182
// MI455X (gfx1250) — matrix operand order repaired and checked by reading, not compiled, not run on hardware
//
#include <hip/hip_runtime.h>
#include <hip/hip_bf16.h>
#include <stdint.h>

// ---------------------------------------------------------------------------
// RuleRNN (GRU + relation attention), B=8192 D=512 NUM_HOP=8 2R=4000
// All matmuls: C[M,N] = A[M,K] * W[N,K]^T (relation pre-transposed once).
// bf16 WMMA (v_wmma_f32_16x16x32_bf16) + async global->LDS double buffering.
// ---------------------------------------------------------------------------

typedef __attribute__((ext_vector_type(16))) __bf16 v16bf;
typedef __attribute__((ext_vector_type(8)))  float  v8f;

#define BATCH 8192
#define DIM   512
#define NHOP  8
#define NREL  4000      // 2 * NUM_REL
#define NRELP 4096      // padded to block-tile multiple (zero rows appended)
#define G3    1536      // 3 * DIM

__device__ __forceinline__ unsigned short f2bf(float f) {
  union { float f; unsigned u; } c; c.f = f;
  unsigned u = c.u;
  u += 0x7fffu + ((u >> 16) & 1u);   // round-to-nearest-even
  return (unsigned short)(u >> 16);
}

// ------------------------------ small utility kernels ----------------------

__global__ void cvt_f32_bf16(const float* __restrict__ src,
                             unsigned short* __restrict__ dst, int n) {
  int i = blockIdx.x * 256 + threadIdx.x;
  if (i < n) dst[i] = f2bf(src[i]);
}

// relB[r, d] for r < NRELP, zero rows beyond NREL (kills in-loop guards)
__global__ void cvt_rel_padded(const float* __restrict__ rel,
                               unsigned short* __restrict__ relB) {
  int i = blockIdx.x * 256 + threadIdx.x;   // over NRELP*DIM
  int r = i >> 9, d = i & 511;
  float v = (r < NREL) ? rel[(size_t)r * DIM + d] : 0.0f;
  relB[i] = f2bf(v);
}

// relT[d, r] = rel[r, d]   (DIM x NREL bf16 from NREL x DIM f32)
__global__ void transpose_bf16(const float* __restrict__ rel,
                               unsigned short* __restrict__ relT) {
  int i = blockIdx.x * 256 + threadIdx.x;     // over DIM*NREL
  int d = i / NREL;
  int r = i - d * NREL;
  relT[(size_t)d * NREL + r] = f2bf(rel[(size_t)r * DIM + d]);
}

__global__ void bcast_eos_bf16(const float* __restrict__ eos,
                               unsigned short* __restrict__ x) {
  int i = blockIdx.x * 256 + threadIdx.x;     // over BATCH*DIM
  x[i] = f2bf(eos[i & (DIM - 1)]);
}

__global__ void zero_state(float* __restrict__ h,
                           unsigned short* __restrict__ hb) {
  int i = blockIdx.x * 256 + threadIdx.x;     // over BATCH*DIM
  h[i] = 0.0f;
  hb[i] = 0;
}

__global__ void fill_ones(float* __restrict__ m, int n) {
  int i = blockIdx.x * 256 + threadIdx.x;
  if (i < n) m[i] = 1.0f;
}

// ------------------------------ bf16 NT GEMM -------------------------------
// C[M,Npad] = A[M,K] * W[Npad,K]^T (+bias); optional bf16 mirror C16.
// Block: 256 threads = 8 waves, 128(M) x 256(N) C tile; wave tile 64x64 =
// 4x4 WMMA accumulators. K stepped by 32. Double-buffered LDS filled by
// global_load_async_to_lds_b128 (ASYNCcnt) so DMA overlaps WMMA compute.
// No bounds checks: caller guarantees M%128==0, Npad%256==0, K%32==0.

#define LDSS 40   // 32 + 8 pad bf16 elems per LDS row -> 80B, 16B aligned

__device__ __forceinline__ void async_ldsb128(unsigned short* lds_dst,
                                              const unsigned short* gsrc) {
  unsigned loff = (unsigned)(uintptr_t)lds_dst;           // LDS byte offset
  unsigned long long ga = (unsigned long long)(uintptr_t)gsrc;
  asm volatile("global_load_async_to_lds_b128 %0, %1, off"
               :: "v"(loff), "v"(ga) : "memory");
}

__device__ __forceinline__ void wait_asynccnt0() {
  asm volatile("s_wait_asynccnt 0x0" ::: "memory");
}

__global__ __launch_bounds__(256)
void gemm_bf16_nt(const unsigned short* __restrict__ A,
                  const unsigned short* __restrict__ W,
                  const float* __restrict__ bias,
                  float* __restrict__ C, int ldc,
                  unsigned short* __restrict__ C16,
                  int Npad, int K) {
  __shared__ __align__(16) unsigned short sA[2][128 * LDSS];
  __shared__ __align__(16) unsigned short sB[2][256 * LDSS];

  const int tid   = threadIdx.x;
  const int wave  = tid >> 5;
  const int lane  = tid & 31;
  const int waveM = wave & 1;        // 2 waves in M (64 each)
  const int waveN = wave >> 1;       // 4 waves in N (64 each)
  const int hi    = lane >> 4;       // K-half select (per ISA frag layout)
  const int l15   = lane & 15;       // row (A) / col (B,C) within fragment
  const int m0    = blockIdx.y * 128;
  const int n0    = blockIdx.x * 256;

  // issue async DMA for one 128x32 (A) + 256x32 (B) bf16 tile pair
  auto stage = [&](int buf, int kk) {
#pragma unroll
    for (int c = 0; c < 2; ++c) {          // A: 512 16B-chunks / 256 thr
      int id = tid + c * 256;
      int row = id >> 2, col = (id & 3) << 3;
      async_ldsb128(&sA[buf][row * LDSS + col],
                    A + (size_t)(m0 + row) * K + kk + col);
    }
#pragma unroll
    for (int c = 0; c < 4; ++c) {          // B: 1024 16B-chunks / 256 thr
      int id = tid + c * 256;
      int row = id >> 2, col = (id & 3) << 3;
      async_ldsb128(&sB[buf][row * LDSS + col],
                    W + (size_t)(n0 + row) * K + kk + col);
    }
  };

  v8f acc[4][4] = {};

  stage(0, 0);
  int buf = 0;
  for (int kk = 0; kk < K; kk += 32, buf ^= 1) {
    wait_asynccnt0();        // my async copies into sX[buf] landed
    __syncthreads();         // everyone's copies landed
    if (kk + 32 < K) stage(buf ^ 1, kk + 32);   // overlap next DMA w/ compute

    union Frag { uint4 q[2]; v16bf v; };
    Frag fa[4], fb[4];
#pragma unroll
    for (int i = 0; i < 4; ++i) {
      const unsigned short* p =
          &sA[buf][(waveM * 64 + i * 16 + l15) * LDSS + hi * 8];
      fa[i].q[0] = *(const uint4*)p;          // K = 8*hi .. 8*hi+7
      fa[i].q[1] = *(const uint4*)(p + 16);   // K = 8*hi+16 .. 8*hi+23
    }
#pragma unroll
    for (int j = 0; j < 4; ++j) {
      const unsigned short* p =
          &sB[buf][(waveN * 64 + j * 16 + l15) * LDSS + hi * 8];
      fb[j].q[0] = *(const uint4*)p;          // K = 8*hi .. 8*hi+7
      fb[j].q[1] = *(const uint4*)(p + 16);   // K = 8*hi+16 .. 8*hi+23
    }

#pragma unroll
    for (int i = 0; i < 4; ++i)
#pragma unroll
      for (int j = 0; j < 4; ++j)
        acc[i][j] = __builtin_amdgcn_wmma_f32_16x16x32_bf16(
            false, fa[i].v, false, fb[j].v, (short)0, acc[i][j],
            false, false);
    __syncthreads();         // done reading sX[buf] before it is refilled
  }

  // epilogue: lane l15 owns column n, VGPR r owns row r + 8*hi
#pragma unroll
  for (int j = 0; j < 4; ++j) {
    int n = n0 + waveN * 64 + j * 16 + l15;
    float bv = bias ? bias[n] : 0.0f;
#pragma unroll
    for (int i = 0; i < 4; ++i) {
#pragma unroll
      for (int r = 0; r < 8; ++r) {
        int m = m0 + waveM * 64 + i * 16 + hi * 8 + r;
        float val = acc[i][j][r] + bv;
        C[(size_t)m * ldc + n] = val;
        if (C16) C16[(size_t)m * Npad + n] = f2bf(val);
      }
    }
  }
}

// ------------------------------ softmax (row-wise) -------------------------
// One block per row of score[BATCH, NRELP] (pad columns ignored); bf16 out.

__global__ __launch_bounds__(256)
void softmax_rows(const float* __restrict__ S,
                  unsigned short* __restrict__ Att) {
  __shared__ float red[256];
  const int b = blockIdx.x;
  const float* row = S + (size_t)b * NRELP;

  float v[16];
  int cnt = 0;
  float mx = -3.4e38f;
  for (int i = threadIdx.x; i < NREL; i += 256) {
    float x = row[i];
    v[cnt++] = x;
    mx = fmaxf(mx, x);
  }
  red[threadIdx.x] = mx;
  __syncthreads();
  for (int s = 128; s > 0; s >>= 1) {
    if (threadIdx.x < s)
      red[threadIdx.x] = fmaxf(red[threadIdx.x], red[threadIdx.x + s]);
    __syncthreads();
  }
  mx = red[0];
  __syncthreads();

  float sum = 0.0f;
  for (int c = 0; c < cnt; ++c) {
    v[c] = __expf(v[c] - mx);     // TAU == 1.0
    sum += v[c];
  }
  red[threadIdx.x] = sum;
  __syncthreads();
  for (int s = 128; s > 0; s >>= 1) {
    if (threadIdx.x < s) red[threadIdx.x] += red[threadIdx.x + s];
    __syncthreads();
  }
  float inv = 1.0f / red[0];

  unsigned short* arow = Att + (size_t)b * NREL;
  cnt = 0;
  for (int i = threadIdx.x; i < NREL; i += 256)
    arow[i] = f2bf(v[cnt++] * inv);
}

// ------------------------------ GRU pointwise ------------------------------
// gi/gh laid out [BATCH, 3*DIM] with gate order (r, z, n).

__global__ __launch_bounds__(256)
void gru_pointwise(const float* __restrict__ gi, const float* __restrict__ gh,
                   float* __restrict__ h, unsigned short* __restrict__ hb) {
  size_t idx = (size_t)blockIdx.x * 256 + threadIdx.x;   // over BATCH*DIM
  size_t b = idx >> 9;
  int    d = (int)(idx & 511);
  size_t base = b * G3 + d;

  float ir = gi[base], iz = gi[base + DIM], in = gi[base + 2 * DIM];
  float hr = gh[base], hz = gh[base + DIM], hn = gh[base + 2 * DIM];

  float r = 1.0f / (1.0f + __expf(-(ir + hr)));
  float z = 1.0f / (1.0f + __expf(-(iz + hz)));
  float n = tanhf(in + r * hn);
  float hnew = (1.0f - z) * n + z * h[idx];

  h[idx]  = hnew;
  hb[idx] = f2bf(hnew);
}

// ------------------------------ launcher -----------------------------------

extern "C" void kernel_launch(void* const* d_in, const int* in_sizes, int n_in,
                              void* d_out, int out_size, void* d_ws,
                              size_t ws_size, hipStream_t stream) {
  const float* query = (const float*)d_in[0];   // [B, D]
  const float* Wih   = (const float*)d_in[1];   // [3D, D]
  const float* Whh   = (const float*)d_in[2];   // [3D, D]
  const float* bih   = (const float*)d_in[3];   // [3D]
  const float* bhh   = (const float*)d_in[4];   // [3D]
  const float* rel   = (const float*)d_in[5];   // [NREL, D]
  const float* eos   = (const float*)d_in[6];   // [1, D]
  float* out = (float*)d_out;                   // subgoals [B,8,D] ++ masks [B,8]

  // ---- workspace carve (256B aligned) ----
  char* p = (char*)d_ws;
  auto carve = [&](size_t bytes) -> char* {
    char* r = p;
    p += (bytes + 255) & ~(size_t)255;
    return r;
  };
  unsigned short* relB = (unsigned short*)carve((size_t)NRELP * DIM * 2);
  unsigned short* relT = (unsigned short*)carve((size_t)DIM * NREL * 2);
  unsigned short* WihB = (unsigned short*)carve((size_t)G3 * DIM * 2);
  unsigned short* WhhB = (unsigned short*)carve((size_t)G3 * DIM * 2);
  float*          hF   = (float*)carve((size_t)BATCH * DIM * 4);
  unsigned short* hB   = (unsigned short*)carve((size_t)BATCH * DIM * 2);
  unsigned short* xB   = (unsigned short*)carve((size_t)BATCH * DIM * 2);
  unsigned short* attB = (unsigned short*)carve((size_t)BATCH * NREL * 2);
  // big region: score (fp32 B x NRELP) overlays gi+gh (fp32 B x 3D each);
  // lifetimes within a hop are disjoint.
  char* big = carve((size_t)BATCH * NRELP * 4);
  float* score = (float*)big;
  float* giF   = (float*)big;
  float* ghF   = (float*)(big + (size_t)BATCH * G3 * 4);

  const dim3 blk(256);
  const int gridBD = (BATCH * DIM) / 256;             // 16384
  const dim3 gGates(G3 / 256, BATCH / 128);           // 6 x 64
  const dim3 gScore(NRELP / 256, BATCH / 128);        // 16 x 64
  const dim3 gSub(DIM / 256, BATCH / 128);            // 2 x 64

  // ---- one-time conversions ----
  cvt_rel_padded<<<(NRELP * DIM) / 256, blk, 0, stream>>>(rel, relB);
  transpose_bf16<<<(DIM * NREL) / 256, blk, 0, stream>>>(rel, relT);
  cvt_f32_bf16<<<(G3 * DIM) / 256, blk, 0, stream>>>(Wih, WihB, G3 * DIM);
  cvt_f32_bf16<<<(G3 * DIM) / 256, blk, 0, stream>>>(Whh, WhhB, G3 * DIM);
  cvt_f32_bf16<<<gridBD, blk, 0, stream>>>(query, xB, BATCH * DIM);
  zero_state<<<gridBD, blk, 0, stream>>>(hF, hB);

  // ---- GRU cell: x in xB, h in hF/hB ----
  auto gru_cell = [&]() {
    gemm_bf16_nt<<<gGates, blk, 0, stream>>>(xB, WihB, bih, giF, G3, nullptr,
                                             G3, DIM);
    gemm_bf16_nt<<<gGates, blk, 0, stream>>>(hB, WhhB, bhh, ghF, G3, nullptr,
                                             G3, DIM);
    gru_pointwise<<<gridBD, blk, 0, stream>>>(giF, ghF, hF, hB);
  };

  // cell 1: x = query, h = 0
  gru_cell();
  // cell 2: x = eos broadcast
  bcast_eos_bf16<<<gridBD, blk, 0, stream>>>(eos, xB);
  gru_cell();

  // ---- hop loop (sequential dependency via stream order) ----
  for (int hop = 0; hop < NHOP; ++hop) {
    // score = h @ rel^T   [B, NRELP] (pad cols are exact zeros)
    gemm_bf16_nt<<<gScore, blk, 0, stream>>>(hB, relB, nullptr, score, NRELP,
                                             nullptr, NRELP, DIM);
    softmax_rows<<<BATCH, blk, 0, stream>>>(score, attB);
    // subgoal = att @ rel = att @ relT^T; fp32 -> d_out, bf16 -> xB
    gemm_bf16_nt<<<gSub, blk, 0, stream>>>(attB, relT, nullptr,
                                           out + (size_t)hop * DIM,
                                           NHOP * DIM, xB, DIM, NREL);
    gru_cell();
  }

  // masks = ones [B, NHOP], appended after subgoals
  fill_ones<<<(BATCH * NHOP) / 256, blk, 0, stream>>>(
      out + (size_t)BATCH * NHOP * DIM, BATCH * NHOP);
}
